// LstmCell_10101763080406
// MI455X (gfx1250) — compile-verified, matrix operand order checked by reading, not run on hardware
//
#include <hip/hip_runtime.h>
#include <hip/hip_bf16.h>

typedef __attribute__((ext_vector_type(16))) __bf16 v16bf;
typedef __attribute__((ext_vector_type(8)))  __bf16 v8bf;
typedef __attribute__((ext_vector_type(4)))  __bf16 v4bf;
typedef __attribute__((ext_vector_type(8)))  float  v8f;
typedef int v4i __attribute__((vector_size(16)));   // 128-bit payload type

#define BATCH   8192
#define DIM     1024      // STATE_DIM == INPUT_DIM
#define KTOT    2048      // concatenated K (x then h)
#define NGATE   4
#define BM      128       // rows per workgroup
#define BN      32        // cols per gate per workgroup
#define BK      32        // K per step (== WMMA K)
#define KSTEPS  (KTOT / BK)   // 64
#define LDSPAD  40        // padded row stride (elements)

#define AS1 __attribute__((address_space(1)))
#define AS3 __attribute__((address_space(3)))

typedef AS3 __bf16 lds_bf;
typedef AS3 v8bf   lds_v8bf;

#if defined(__has_builtin)
#if __has_builtin(__builtin_amdgcn_global_load_async_to_lds_b128)
#define USE_ASYNC_COPY 1
#endif
#endif

__device__ __forceinline__ float sigmoidf_fast(float z) {
    return 1.0f / (1.0f + __expf(-z));
}

// 16-byte global(bf16) -> LDS copy; LDS side already addrspace(3).
__device__ __forceinline__ void copy16(const __bf16* g, lds_bf* l) {
#ifdef USE_ASYNC_COPY
    __builtin_amdgcn_global_load_async_to_lds_b128(
        (AS1 v4i*)(v4i*)g, (AS3 v4i*)l, 0, 0);
#else
    *(lds_v8bf*)l = *(const v8bf*)g;
#endif
}

__device__ __forceinline__ void wait_async_copies() {
#ifdef USE_ASYNC_COPY
#if __has_builtin(__builtin_amdgcn_s_wait_asynccnt)
    __builtin_amdgcn_s_wait_asynccnt(0);
#else
    asm volatile("s_wait_asynccnt 0" ::: "memory");
#endif
#endif
}

// ---------------------------------------------------------------------------
// Phase 1a: Acat[row][k] = bf16( k<1024 ? x[row][k] : prevh[row][k-1024] )
// ---------------------------------------------------------------------------
__global__ __launch_bounds__(256)
void convertA_kernel(const float* __restrict__ x,
                     const float* __restrict__ prevh,
                     __bf16* __restrict__ Acat)
{
    const int t = blockIdx.x * 256 + threadIdx.x;   // 1,048,576 threads
    const int TOT4 = BATCH * KTOT / 4;              // 4,194,304 float4-equivalents
#pragma unroll
    for (int i = 0; i < 4; ++i) {
        const int idx = t + i * (TOT4 / 4);
        const int row = idx >> 9;                   // KTOT/4 = 512 float4 per row
        const int c4  = idx & 511;
        const int k   = c4 * 4;
        const float* src = (k < DIM) ? (x + (size_t)row * DIM + k)
                                     : (prevh + (size_t)row * DIM + (k - DIM));
        const float4 v = *(const float4*)src;
        v4bf bv = { (__bf16)v.x, (__bf16)v.y, (__bf16)v.z, (__bf16)v.w };
        *(v4bf*)(Acat + (size_t)row * KTOT + k) = bv;
    }
}

// ---------------------------------------------------------------------------
// Phase 1b: Wcat[n][k] = bf16( k<1024 ? Wx[k][n] : Wh[k-1024][n] )  (transpose)
// ---------------------------------------------------------------------------
__global__ __launch_bounds__(256)
void convertW_kernel(const float* __restrict__ Wx,
                     const float* __restrict__ Wh,
                     __bf16* __restrict__ Wcat)
{
    __shared__ __bf16 T[32][33];
    const int n0 = blockIdx.x * 32;        // 0..4095
    const int k0 = blockIdx.y * 32;        // 0..2047
    const int tx = threadIdx.x & 31;
    const int ty = threadIdx.x >> 5;       // 0..7

    const float* src = (k0 < DIM) ? Wx : Wh;
    const int kb = (k0 < DIM) ? k0 : (k0 - DIM);

#pragma unroll
    for (int i = 0; i < 4; ++i) {
        const int kk = ty * 4 + i;
        T[tx][kk] = (__bf16)src[(size_t)(kb + kk) * (NGATE * DIM) + n0 + tx];
    }
    __syncthreads();
#pragma unroll
    for (int i = 0; i < 4; ++i) {
        const int rr = ty * 4 + i;
        Wcat[(size_t)(n0 + rr) * KTOT + k0 + tx] = T[rr][tx];
    }
}

// ---------------------------------------------------------------------------
// Phase 2: fused GEMM (bf16 WMMA, f32 accum) + LSTM gate epilogue
// ---------------------------------------------------------------------------
__global__ __launch_bounds__(256)
void lstm_gemm_kernel(const __bf16* __restrict__ Acat,
                      const __bf16* __restrict__ Wcat,
                      const float* __restrict__ prevc,
                      const float* __restrict__ bx,
                      float* __restrict__ nexth,
                      float* __restrict__ nextc)
{
    __shared__ __bf16 As[2][BM][LDSPAD];            // 128 x 32 (padded)
    __shared__ __bf16 Bs[2][NGATE * BN][LDSPAD];    // 128 x 32 (padded), gate-major

    const int tid   = threadIdx.x;
    const int lane  = tid & 31;
    const int wid   = tid >> 5;
    const int waveM = wid & 3;
    const int waveN = wid >> 2;
    const int lmod  = lane & 15;
    const int lhalf = lane >> 4;
    const int koff  = lhalf * 8;

    const int blockM = blockIdx.x * BM;
    const int n0     = blockIdx.y * BN;

    // ---- per-thread copy assignments (fixed; only k advances) ----
    const __bf16* srcA[2];
    const __bf16* srcB[2];
    lds_bf*       dA0[2];    // destinations in buffer 0
    lds_bf*       dA1[2];    // destinations in buffer 1
    lds_bf*       dB0[2];
    lds_bf*       dB1[2];
#pragma unroll
    for (int i = 0; i < 2; ++i) {
        const int c   = tid + i * 256;
        const int row = c >> 2;
        const int cq  = c & 3;
        srcA[i] = Acat + (size_t)(blockM + row) * KTOT + cq * 8;
        dA0[i]  = (lds_bf*)&As[0][row][cq * 8];
        dA1[i]  = (lds_bf*)&As[1][row][cq * 8];

        const int g  = c >> 7;
        const int r  = (c >> 2) & 31;
        srcB[i] = Wcat + (size_t)(g * DIM + n0 + r) * KTOT + cq * 8;
        dB0[i]  = (lds_bf*)&Bs[0][g * BN + r][cq * 8];
        dB1[i]  = (lds_bf*)&Bs[1][g * BN + r][cq * 8];
    }

    v8f acc[NGATE][2];
#pragma unroll
    for (int g = 0; g < NGATE; ++g)
#pragma unroll
        for (int mi = 0; mi < 2; ++mi)
#pragma unroll
            for (int j = 0; j < 8; ++j)
                acc[g][mi][j] = 0.0f;

    auto issue = [&](lds_bf* const* dA, lds_bf* const* dB) {
#pragma unroll
        for (int i = 0; i < 2; ++i) {
            copy16(srcA[i], dA[i]);
            copy16(srcB[i], dB[i]);
            srcA[i] += BK;
            srcB[i] += BK;
        }
    };

    auto compute = [&](int cbuf) {   // cbuf is a literal at every call site
        v16bf afrag[2];
#pragma unroll
        for (int mi = 0; mi < 2; ++mi) {
            const __bf16* p = &As[cbuf][waveM * 32 + mi * 16 + lmod][koff];
            ((v8bf*)&afrag[mi])[0] = *(const v8bf*)p;
            ((v8bf*)&afrag[mi])[1] = *(const v8bf*)(p + 16);
        }
        v16bf bfrag[NGATE];
#pragma unroll
        for (int g = 0; g < NGATE; ++g) {
            const __bf16* q = &Bs[cbuf][g * BN + waveN * 16 + lmod][koff];
            ((v8bf*)&bfrag[g])[0] = *(const v8bf*)q;
            ((v8bf*)&bfrag[g])[1] = *(const v8bf*)(q + 16);
        }
#pragma unroll
        for (int g = 0; g < NGATE; ++g) {
            acc[g][0] = __builtin_amdgcn_wmma_f32_16x16x32_bf16(
                false, afrag[0], false, bfrag[g], (short)0, acc[g][0], false, false);
            acc[g][1] = __builtin_amdgcn_wmma_f32_16x16x32_bf16(
                false, afrag[1], false, bfrag[g], (short)0, acc[g][1], false, false);
        }
    };

    // prologue: tile 0 -> buffer 0
    issue(dA0, dB0);
    wait_async_copies();
    __syncthreads();

    // K-loop unrolled by 2: buffer indices are compile-time constants.
#pragma unroll 1
    for (int kt = 0; kt < KSTEPS; kt += 2) {
        issue(dA1, dB1);                      // tile kt+1 -> buffer 1
        compute(0);
        wait_async_copies();
        __syncthreads();

        if (kt + 2 < KSTEPS) issue(dA0, dB0); // tile kt+2 -> buffer 0
        compute(1);
        wait_async_copies();
        __syncthreads();
    }

    // ---- fused LSTM epilogue ----
    const int col = n0 + waveN * 16 + lmod;
    const float bi = bx[0 * DIM + col];
    const float bf = bx[1 * DIM + col];
    const float bo = bx[2 * DIM + col];
    const float bg = bx[3 * DIM + col];

#pragma unroll
    for (int mi = 0; mi < 2; ++mi) {
#pragma unroll
        for (int j = 0; j < 8; ++j) {
            const int row = blockM + waveM * 32 + mi * 16 + lhalf * 8 + j;
            const size_t off = (size_t)row * DIM + col;

            const float gi = sigmoidf_fast(acc[0][mi][j] + bi);
            const float gf = sigmoidf_fast(acc[1][mi][j] + bf);
            const float go = sigmoidf_fast(acc[2][mi][j] + bo);
            const float gg = tanhf(acc[3][mi][j] + bg);

            const float pc = prevc[off];
            const float nc = pc * gf + gg * gi;
            const float nh = tanhf(nc) * go;

            nextc[off] = nc;
            nexth[off] = nh;
        }
    }
}

// ---------------------------------------------------------------------------
// Fallback: fully fused single kernel, used when workspace is too small.
// ---------------------------------------------------------------------------
__global__ __launch_bounds__(256)
void lstm_fused_fallback(const float* __restrict__ x,
                         const float* __restrict__ prevh,
                         const float* __restrict__ prevc,
                         const float* __restrict__ Wx,
                         const float* __restrict__ bx,
                         const float* __restrict__ Wh,
                         float* __restrict__ nexth,
                         float* __restrict__ nextc)
{
    __shared__ __bf16 As[2][BM][LDSPAD];
    __shared__ __bf16 Bs[2][NGATE][BN][LDSPAD];

    const int tid   = threadIdx.x;
    const int lane  = tid & 31;
    const int wid   = tid >> 5;
    const int waveM = wid & 3;
    const int waveN = wid >> 2;
    const int lmod  = lane & 15;
    const int lhalf = lane >> 4;
    const int koff  = lhalf * 8;

    const int blockM = blockIdx.x * BM;
    const int n0     = blockIdx.y * BN;

    v8f acc[NGATE][2];
#pragma unroll
    for (int g = 0; g < NGATE; ++g)
#pragma unroll
        for (int mi = 0; mi < 2; ++mi)
#pragma unroll
            for (int j = 0; j < 8; ++j)
                acc[g][mi][j] = 0.0f;

    auto load_tile = [&](int kt, int b) {
        const float* Asrc; const float* Wsrc; int k0;
        if (kt < KSTEPS / 2) { Asrc = x;     Wsrc = Wx; k0 = kt * BK; }
        else                 { Asrc = prevh; Wsrc = Wh; k0 = (kt - KSTEPS / 2) * BK; }
#pragma unroll
        for (int i = 0; i < 4; ++i) {
            const int idx = tid + i * 256;
            const int row = idx >> 3;
            const int c4  = idx & 7;
            const float4 v = *(const float4*)(Asrc + (size_t)(blockM + row) * DIM + k0 + c4 * 4);
            v4bf bv = { (__bf16)v.x, (__bf16)v.y, (__bf16)v.z, (__bf16)v.w };
            *(v4bf*)&As[b][row][c4 * 4] = bv;
        }
        const int k  = tid >> 3;
        const int n4 = tid & 7;
#pragma unroll
        for (int g = 0; g < NGATE; ++g) {
            const float4 w = *(const float4*)(Wsrc + (size_t)(k0 + k) * (NGATE * DIM)
                                              + g * DIM + n0 + n4 * 4);
            Bs[b][g][n4 * 4 + 0][k] = (__bf16)w.x;
            Bs[b][g][n4 * 4 + 1][k] = (__bf16)w.y;
            Bs[b][g][n4 * 4 + 2][k] = (__bf16)w.z;
            Bs[b][g][n4 * 4 + 3][k] = (__bf16)w.w;
        }
    };

    load_tile(0, 0);
    __syncthreads();

    int buf = 0;
#pragma unroll 1
    for (int kt = 0; kt < KSTEPS; ++kt) {
        if (kt + 1 < KSTEPS) load_tile(kt + 1, buf ^ 1);

        v16bf afrag[2];
#pragma unroll
        for (int mi = 0; mi < 2; ++mi) {
            const __bf16* p = &As[buf][waveM * 32 + mi * 16 + lmod][koff];
            ((v8bf*)&afrag[mi])[0] = *(const v8bf*)p;
            ((v8bf*)&afrag[mi])[1] = *(const v8bf*)(p + 16);
        }
#pragma unroll
        for (int g = 0; g < NGATE; ++g) {
            v16bf bfr;
            const __bf16* q = &Bs[buf][g][waveN * 16 + lmod][koff];
            ((v8bf*)&bfr)[0] = *(const v8bf*)q;
            ((v8bf*)&bfr)[1] = *(const v8bf*)(q + 16);
            acc[g][0] = __builtin_amdgcn_wmma_f32_16x16x32_bf16(
                false, afrag[0], false, bfr, (short)0, acc[g][0], false, false);
            acc[g][1] = __builtin_amdgcn_wmma_f32_16x16x32_bf16(
                false, afrag[1], false, bfr, (short)0, acc[g][1], false, false);
        }
        __syncthreads();
        buf ^= 1;
    }

    const int col = n0 + waveN * 16 + lmod;
    const float bi = bx[0 * DIM + col];
    const float bf = bx[1 * DIM + col];
    const float bo = bx[2 * DIM + col];
    const float bg = bx[3 * DIM + col];
#pragma unroll
    for (int mi = 0; mi < 2; ++mi) {
#pragma unroll
        for (int j = 0; j < 8; ++j) {
            const int row = blockM + waveM * 32 + mi * 16 + lhalf * 8 + j;
            const size_t off = (size_t)row * DIM + col;
            const float gi = sigmoidf_fast(acc[0][mi][j] + bi);
            const float gf = sigmoidf_fast(acc[1][mi][j] + bf);
            const float go = sigmoidf_fast(acc[2][mi][j] + bo);
            const float gg = tanhf(acc[3][mi][j] + bg);
            const float pc = prevc[off];
            const float nc = pc * gf + gg * gi;
            const float nh = tanhf(nc) * go;
            nextc[off] = nc;
            nexth[off] = nh;
        }
    }
}

extern "C" void kernel_launch(void* const* d_in, const int* in_sizes, int n_in,
                              void* d_out, int out_size, void* d_ws, size_t ws_size,
                              hipStream_t stream) {
    (void)in_sizes; (void)n_in; (void)out_size;
    const float* x     = (const float*)d_in[0];
    const float* prevh = (const float*)d_in[1];
    const float* prevc = (const float*)d_in[2];
    const float* Wx    = (const float*)d_in[3];
    const float* bx    = (const float*)d_in[4];
    const float* Wh    = (const float*)d_in[5];

    float* out   = (float*)d_out;
    float* nexth = out;
    float* nextc = out + (size_t)BATCH * DIM;

    const size_t needA = (size_t)BATCH * KTOT * sizeof(__bf16);         // 32 MB
    const size_t needW = (size_t)(NGATE * DIM) * KTOT * sizeof(__bf16); // 16 MB

    if (ws_size >= needA + needW) {
        __bf16* Acat = (__bf16*)d_ws;
        __bf16* Wcat = (__bf16*)((char*)d_ws + needA);

        convertA_kernel<<<4096, 256, 0, stream>>>(x, prevh, Acat);
        convertW_kernel<<<dim3((NGATE * DIM) / 32, KTOT / 32), 256, 0, stream>>>(Wx, Wh, Wcat);

        dim3 grid(BATCH / BM, DIM / BN);   // 64 x 32
        lstm_gemm_kernel<<<grid, 256, 0, stream>>>(Acat, Wcat, prevc, bx, nexth, nextc);
    } else {
        dim3 grid(BATCH / BM, DIM / BN);
        lstm_fused_fallback<<<grid, 256, 0, stream>>>(x, prevh, prevc, Wx, bx, Wh,
                                                      nexth, nextc);
    }
}
